// attention_block_50190987821768
// MI455X (gfx1250) — compile-verified, matrix operand order checked by reading, not run on hardware
//
#include <hip/hip_runtime.h>

typedef __attribute__((ext_vector_type(16))) __bf16 v16bf;
typedef __attribute__((ext_vector_type(8)))  float  v8f;

#define WMMA_BF16(a, b, c) \
  __builtin_amdgcn_wmma_f32_16x16x32_bf16(false, (a), false, (b), (short)0, (c), false, false)

// Load a 16-bit A/B WMMA fragment: two contiguous 16-byte chunks at p and p+16 elems.
// CDNA5 layout: lane L holds K = (L>>4)*8 + {0..7} and (L>>4)*8 + {16..23}.
__device__ __forceinline__ v16bf load_frag(const __bf16* p) {
  union { v16bf v; uint4 q[2]; } u;
  u.q[0] = *reinterpret_cast<const uint4*>(p);
  u.q[1] = *reinterpret_cast<const uint4*>(p + 16);
  return u.v;
}

__device__ __forceinline__ float rmax16(float v) {
  v = fmaxf(v, __shfl_xor(v, 1, 32));
  v = fmaxf(v, __shfl_xor(v, 2, 32));
  v = fmaxf(v, __shfl_xor(v, 4, 32));
  v = fmaxf(v, __shfl_xor(v, 8, 32));
  return v;
}
__device__ __forceinline__ float rsum16(float v) {
  v += __shfl_xor(v, 1, 32);
  v += __shfl_xor(v, 2, 32);
  v += __shfl_xor(v, 4, 32);
  v += __shfl_xor(v, 8, 32);
  return v;
}

// ---------------- elementwise cast / transpose kernels ----------------

__global__ void cvt_f32_bf16(const float* __restrict__ s, __bf16* __restrict__ d, int n) {
  int i = blockIdx.x * blockDim.x + threadIdx.x;
  if (i < n) d[i] = (__bf16)s[i];
}

// src is K x N row-major fp32; dst is N x K row-major bf16 (transposed).
__global__ void transpose_cvt(const float* __restrict__ s, __bf16* __restrict__ d,
                              int K, int N) {
  int i = blockIdx.x * blockDim.x + threadIdx.x;
  if (i < K * N) {
    int n = i / K;
    int k = i - n * K;
    d[i] = (__bf16)s[(size_t)k * N + n];
  }
}

// Build Vt[b][h][d][n] (bf16) from qkv[b*1024+n][2048 + h*64 + d].
__global__ void build_vt(const __bf16* __restrict__ qkv, __bf16* __restrict__ vt) {
  int i = blockIdx.x * blockDim.x + threadIdx.x;      // 4*16*64*1024 total
  int n = i & 1023;
  int d = (i >> 10) & 63;
  int h = (i >> 16) & 15;
  int b = i >> 20;
  vt[i] = qkv[((size_t)(b * 1024 + n)) * 3072 + 2048 + h * 64 + d];
}

// ---------------- WMMA GEMM: C[M,N] = A[M,K] @ Bt[N,K]^T + bias ----------------
// Each wave: 16 rows x 64 cols (4 accumulator tiles). Ping-pong double buffer,
// manually unrolled by 2 so no register rotation copies are generated.
// Requires K % 64 == 0.

__global__ __launch_bounds__(256) void gemm_bf16(
    const __bf16* __restrict__ A, const __bf16* __restrict__ Bt,
    const float* __restrict__ bias,
    float* __restrict__ outf, __bf16* __restrict__ outb,
    int M, int N, int K) {
  const int lane = threadIdx.x & 31;
  const int wave = threadIdx.x >> 5;
  const int lr = lane & 15;        // row/col within tile
  const int lh = lane >> 4;        // half-wave select
  const int row0 = blockIdx.y * 128 + wave * 16;
  const int col0 = blockIdx.x * 64;

  const __bf16* ap = A + (size_t)(row0 + lr) * K + lh * 8;
  const __bf16* bp[4];
#pragma unroll
  for (int t = 0; t < 4; ++t)
    bp[t] = Bt + (size_t)(col0 + t * 16 + lr) * K + lh * 8;

  v8f acc[4];
  v8f z = {};
#pragma unroll
  for (int t = 0; t < 4; ++t) acc[t] = z;

  // prologue: fill both buffers (chunks 0 and 32)
  v16bf a0 = load_frag(ap);
  v16bf a1 = load_frag(ap + 32);
  v16bf b0[4], b1[4];
#pragma unroll
  for (int t = 0; t < 4; ++t) b0[t] = load_frag(bp[t]);
#pragma unroll
  for (int t = 0; t < 4; ++t) b1[t] = load_frag(bp[t] + 32);

  // steady state: compute chunk kk / kk+32, prefetch kk+64 / kk+96
  int kk = 0;
  for (; kk < K - 64; kk += 64) {
#pragma unroll
    for (int t = 0; t < 4; ++t) acc[t] = WMMA_BF16(a0, b0[t], acc[t]);
    a0 = load_frag(ap + kk + 64);
#pragma unroll
    for (int t = 0; t < 4; ++t) b0[t] = load_frag(bp[t] + kk + 64);

#pragma unroll
    for (int t = 0; t < 4; ++t) acc[t] = WMMA_BF16(a1, b1[t], acc[t]);
    a1 = load_frag(ap + kk + 96);
#pragma unroll
    for (int t = 0; t < 4; ++t) b1[t] = load_frag(bp[t] + kk + 96);
  }

  // tail: last two chunks, no prefetch
#pragma unroll
  for (int t = 0; t < 4; ++t) acc[t] = WMMA_BF16(a0, b0[t], acc[t]);
#pragma unroll
  for (int t = 0; t < 4; ++t) acc[t] = WMMA_BF16(a1, b1[t], acc[t]);

#pragma unroll
  for (int t = 0; t < 4; ++t) {
    const int col = col0 + t * 16 + lr;
    const float bi = bias[col];
#pragma unroll
    for (int r = 0; r < 8; ++r) {
      const int row = row0 + r + lh * 8;
      const float v = acc[t][r] + bi;
      if (outf) outf[(size_t)row * N + col] = v;
      else      outb[(size_t)row * N + col] = (__bf16)v;
    }
  }
}

// ---------------- Fused flash attention ----------------
// grid.x = b*16 + h (64), grid.y = q-tile group (8). Block = 8 waves;
// each wave owns one 16-row Q tile, streams KV in chunks of 32 rows.

__global__ __launch_bounds__(256) void attn_fused(
    const __bf16* __restrict__ qkv,   // [4096, 3072] bf16 (K|Q|V)
    const __bf16* __restrict__ vt,    // [b][h][64][1024] bf16
    __bf16* __restrict__ o) {         // [4096, 1024] bf16
  __shared__ __align__(16) __bf16 pbuf[8][16 * 32];

  const int lane = threadIdx.x & 31;
  const int wave = threadIdx.x >> 5;
  const int lr = lane & 15;
  const int lh = lane >> 4;
  const int b = blockIdx.x >> 4;
  const int h = blockIdx.x & 15;
  const int qbase = (blockIdx.y * 8 + wave) * 16;

  // Q fragments (16 x 64 -> two 16x32 A-frags)
  const __bf16* qp =
      qkv + ((size_t)(b * 1024 + qbase + lr)) * 3072 + 1024 + h * 64 + lh * 8;
  const v16bf q0 = load_frag(qp);
  const v16bf q1 = load_frag(qp + 32);

  const __bf16* kbase = qkv + (size_t)b * 1024 * 3072 + h * 64 + lh * 8;
  const __bf16* vbase = vt + ((size_t)((b * 16 + h) * 64 + lr)) * 1024 + lh * 8;

  v8f acc[4];
  v8f z = {};
#pragma unroll
  for (int t = 0; t < 4; ++t) acc[t] = z;
  float m[8], l[8];
#pragma unroll
  for (int r = 0; r < 8; ++r) { m[r] = -1e30f; l[r] = 0.0f; }

  for (int j = 0; j < 1024; j += 32) {
    // ---- scores: two 16x16 tiles over KV rows [j, j+32)
    const __bf16* kp0 = kbase + (size_t)(j + lr) * 3072;
    const __bf16* kp1 = kbase + (size_t)(j + 16 + lr) * 3072;
    v8f s0 = WMMA_BF16(q0, load_frag(kp0), z);
    s0 = WMMA_BF16(q1, load_frag(kp0 + 32), s0);
    v8f s1 = WMMA_BF16(q0, load_frag(kp1), z);
    s1 = WMMA_BF16(q1, load_frag(kp1 + 32), s1);

    // ---- issue V loads NOW; the softmax VALU block below hides their latency
    v16bf vfrag[4];
#pragma unroll
    for (int dt = 0; dt < 4; ++dt)
      vfrag[dt] = load_frag(vbase + (size_t)dt * 16 * 1024 + j);

    // ---- online softmax update (row M = r + 8*lh lives in VGPR r, half lh)
    float alpha[8];
#pragma unroll
    for (int r = 0; r < 8; ++r) {
      float a0 = s0[r] * 0.125f;         // 1/sqrt(64)
      float a1 = s1[r] * 0.125f;
      float mt = rmax16(fmaxf(a0, a1));
      float mn = fmaxf(m[r], mt);
      float p0 = __expf(a0 - mn);
      float p1 = __expf(a1 - mn);
      float al = __expf(m[r] - mn);
      l[r] = l[r] * al + rsum16(p0 + p1);
      m[r] = mn;
      alpha[r] = al;
      s0[r] = p0;
      s1[r] = p1;
    }
#pragma unroll
    for (int t = 0; t < 4; ++t)
#pragma unroll
      for (int r = 0; r < 8; ++r) acc[t][r] *= alpha[r];

    // ---- repack P (C-layout) -> A-frag layout through per-wave LDS tile
    __bf16* pw = &pbuf[wave][0];
#pragma unroll
    for (int r = 0; r < 8; ++r) {
      const int row = r + lh * 8;
      pw[row * 32 + lr]      = (__bf16)s0[r];
      pw[row * 32 + 16 + lr] = (__bf16)s1[r];
    }
    asm volatile("s_wait_dscnt 0" ::: "memory");
    const v16bf pa = load_frag(&pbuf[wave][lr * 32 + lh * 8]);
    asm volatile("" ::: "memory");

    // ---- O += P @ V  (V chunk 32 x 64, four 16-wide d subtiles)
#pragma unroll
    for (int dt = 0; dt < 4; ++dt)
      acc[dt] = WMMA_BF16(pa, vfrag[dt], acc[dt]);
  }

  // ---- normalize and write out as [b, n, h*64 + d]
#pragma unroll
  for (int r = 0; r < 8; ++r) {
    const float inv = 1.0f / l[r];
    const int row = qbase + r + lh * 8;
#pragma unroll
    for (int dt = 0; dt < 4; ++dt) {
      const int col = h * 64 + dt * 16 + lr;
      o[(size_t)(b * 1024 + row) * 1024 + col] = (__bf16)(acc[dt][r] * inv);
    }
  }
}

// ---------------- host-side launcher ----------------

extern "C" void kernel_launch(void* const* d_in, const int* in_sizes, int n_in,
                              void* d_out, int out_size, void* d_ws, size_t ws_size,
                              hipStream_t stream) {
  const float* x     = (const float*)d_in[0];   // [4,1024,1024]
  const float* W_qkv = (const float*)d_in[1];   // [1024,3072]
  const float* b_qkv = (const float*)d_in[2];   // [3072]
  const float* W_out = (const float*)d_in[3];   // [1024,1024]
  const float* b_out = (const float*)d_in[4];   // [1024]
  float* out = (float*)d_out;                   // [4,1024,1024]

  const size_t M = 4096;   // tokens
  const size_t D = 1024;   // emb
  const size_t N3 = 3072;  // qkv cols

  __bf16* xb    = (__bf16*)d_ws;                 // 4096*1024
  __bf16* wqkvt = xb + M * D;                    // 3072*1024 (N x K)
  __bf16* woutt = wqkvt + N3 * D;                // 1024*1024 (N x K)
  __bf16* qkvb  = woutt + D * D;                 // 4096*3072
  __bf16* vtb   = qkvb + M * N3;                 // 64*64*1024
  __bf16* attnb = vtb + (size_t)64 * 64 * 1024;  // 4096*1024

  // 1) casts / transposes
  cvt_f32_bf16<<<(M * D + 255) / 256, 256, 0, stream>>>(x, xb, (int)(M * D));
  transpose_cvt<<<(D * N3 + 255) / 256, 256, 0, stream>>>(W_qkv, wqkvt, (int)D, (int)N3);
  transpose_cvt<<<(D * D + 255) / 256, 256, 0, stream>>>(W_out, woutt, (int)D, (int)D);

  // 2) QKV projection: [4096,1024] @ [1024,3072] + b_qkv -> bf16
  gemm_bf16<<<dim3(48, 32), 256, 0, stream>>>(xb, wqkvt, b_qkv,
                                              nullptr, qkvb,
                                              (int)M, (int)N3, (int)D);

  // 3) V transpose for PV matmul
  build_vt<<<(64 * 64 * 1024) / 256, 256, 0, stream>>>(qkvb, vtb);

  // 4) fused flash attention
  attn_fused<<<dim3(64, 8), 256, 0, stream>>>(qkvb, vtb, attnb);

  // 5) output projection: [4096,1024] @ [1024,1024] + b_out -> fp32
  gemm_bf16<<<dim3(16, 32), 256, 0, stream>>>(attnb, woutt, b_out,
                                              out, nullptr,
                                              (int)M, (int)D, (int)D);
}
